// EncoderLayer_61753039782317
// MI455X (gfx1250) — matrix operand order repaired and checked by reading, not compiled, not run on hardware
//
#include <hip/hip_runtime.h>

// ---------------------------------------------------------------------------
// Types for CDNA5 WMMA
// ---------------------------------------------------------------------------
typedef __attribute__((ext_vector_type(16))) __bf16 bf16x16;
typedef __attribute__((ext_vector_type(8)))  float  f32x8;

union Frag {
    uint4   u[2];
    bf16x16 v;
};

__device__ __forceinline__ f32x8 wmma_bf16(const bf16x16& a, const bf16x16& b, const f32x8& c) {
    return __builtin_amdgcn_wmma_f32_16x16x32_bf16(false, a, false, b, (short)0, c, false, false);
}

__device__ __forceinline__ unsigned short f2bf(float f) {
    unsigned int u = __builtin_bit_cast(unsigned int, f);
    unsigned int r = ((u >> 16) & 1u) + 0x7fffu;   // round-to-nearest-even
    return (unsigned short)((u + r) >> 16);
}

// ---------------------------------------------------------------------------
// CDNA5 async global->LDS copy (VGPR-bypassing DMA, tracked by ASYNCcnt)
// ---------------------------------------------------------------------------
__device__ __forceinline__ void async_copy16(unsigned lds_off, const void* gaddr) {
    asm volatile("global_load_async_to_lds_b128 %0, %1, off"
                 :: "v"(lds_off), "v"((unsigned long long)gaddr)
                 : "memory");
}
__device__ __forceinline__ void wait_async0() {
    asm volatile("s_wait_asynccnt 0" ::: "memory");
}

#define BB 4
#define SS 1024
#define DD 1024
#define HH 16
#define DK 64
#define FFN 4096
#define MM (BB * SS)

// ---------------------------------------------------------------------------
// Weight transpose + fp32 -> bf16 convert:  out[n][k] = bf16(in[k][n])
// in: [K][N] fp32 row-major. grid = (N/32, K/32), block = (32,8)
// ---------------------------------------------------------------------------
template <int K, int N>
__global__ void transpose_cvt_kernel(const float* __restrict__ in,
                                     unsigned short* __restrict__ out) {
    __shared__ float t[32][33];
    const int k0 = blockIdx.y * 32, n0 = blockIdx.x * 32;
    const int tx = threadIdx.x;
    #pragma unroll
    for (int i = threadIdx.y; i < 32; i += 8)
        t[i][tx] = in[(size_t)(k0 + i) * N + n0 + tx];
    __syncthreads();
    #pragma unroll
    for (int i = threadIdx.y; i < 32; i += 8)
        out[(size_t)(n0 + i) * K + k0 + tx] = f2bf(t[tx][i]);
}

// ---------------------------------------------------------------------------
// LayerNorm (one row of D per block) -> bf16 output
// ---------------------------------------------------------------------------
__global__ __launch_bounds__(256) void ln_bf16_kernel(const float* __restrict__ x,
                                                      const float* __restrict__ g,
                                                      const float* __restrict__ b,
                                                      unsigned short* __restrict__ out) {
    const int row = blockIdx.x;
    const float* xr = x + (size_t)row * DD;
    float s = 0.f, sq = 0.f;
    #pragma unroll
    for (int i = threadIdx.x; i < DD; i += 256) {
        float v = xr[i];
        s += v; sq += v * v;
    }
    __shared__ float red[2][32];
    #pragma unroll
    for (int off = 16; off; off >>= 1) {
        s  += __shfl_down(s,  off, 32);
        sq += __shfl_down(sq, off, 32);
    }
    const int wid = threadIdx.x >> 5, lane = threadIdx.x & 31;
    if (lane == 0) { red[0][wid] = s; red[1][wid] = sq; }
    __syncthreads();
    if (threadIdx.x == 0) {
        float a = 0.f, c = 0.f;
        #pragma unroll
        for (int i = 0; i < 8; i++) { a += red[0][i]; c += red[1][i]; }
        red[0][0] = a; red[1][0] = c;
    }
    __syncthreads();
    const float mean = red[0][0] * (1.0f / DD);
    const float var  = red[1][0] * (1.0f / DD) - mean * mean;
    const float inv  = rsqrtf(var + 1e-6f);
    #pragma unroll
    for (int i = threadIdx.x; i < DD; i += 256) {
        float v = (xr[i] - mean) * inv * g[i] + b[i];
        out[(size_t)row * DD + i] = f2bf(v);
    }
}

// ---------------------------------------------------------------------------
// WMMA bf16 GEMM:  out[M][N] = act( A[M][K] @ BT[N][K]^T + bias ) + res
// block = 256 threads (8 waves), block tile 128x128, wave tile 64x32.
// Ping-pong LDS with ASYNC global->LDS DMA: tile k+1 streams into the other
// buffer while the 8-WMMA burst runs on tile k. One barrier per k-step.
// ---------------------------------------------------------------------------
template <int M, int N, int K, bool HAS_BIAS, bool RELU, bool RES, bool OUT_BF16>
__global__ __launch_bounds__(256) void gemm_bf16_kernel(
    const unsigned short* __restrict__ A,
    const unsigned short* __restrict__ BT,
    const float* __restrict__ bias,
    const float* __restrict__ res,
    void* __restrict__ outp)
{
    // 32 bf16 data + 8 bf16 pad per row -> 80B stride (conflict-free b128 frag reads)
    __shared__ __align__(16) unsigned short As[2][128 * 40];
    __shared__ __align__(16) unsigned short Bs[2][128 * 40];
    constexpr unsigned BUFB = 128 * 40 * 2;   // bytes per buffer

    const int tid  = threadIdx.x;
    const int lane = tid & 31, wid = tid >> 5;
    const int l15  = lane & 15, hi = lane >> 4;
    const int waveM = wid >> 2, waveN = wid & 3;         // 2 x 4 wave grid
    const int mBase = blockIdx.y * 128;
    const int nBase = blockIdx.x * 128;

    // staging geometry: two 16B chunks per thread per matrix per k-step
    const int r0 = tid >> 2,          o0 = (tid & 3) * 8;          // rows 0..63
    const int r1 = (tid + 256) >> 2,  o1 = (tid & 3) * 8;          // rows 64..127
    const unsigned short* Ap0 = A  + (size_t)(mBase + r0) * K + o0;
    const unsigned short* Ap1 = A  + (size_t)(mBase + r1) * K + o1;
    const unsigned short* Bp0 = BT + (size_t)(nBase + r0) * K + o0;
    const unsigned short* Bp1 = BT + (size_t)(nBase + r1) * K + o1;
    const unsigned ldsA0 = (unsigned)(size_t)&As[0][r0 * 40 + o0];
    const unsigned ldsA1 = (unsigned)(size_t)&As[0][r1 * 40 + o1];
    const unsigned ldsB0 = (unsigned)(size_t)&Bs[0][r0 * 40 + o0];
    const unsigned ldsB1 = (unsigned)(size_t)&Bs[0][r1 * 40 + o1];

    f32x8 acc[4][2];
    #pragma unroll
    for (int mf = 0; mf < 4; mf++)
        #pragma unroll
        for (int nf = 0; nf < 2; nf++)
            acc[mf][nf] = (f32x8){0.f,0.f,0.f,0.f,0.f,0.f,0.f,0.f};

    constexpr int NT = K / 32;

    // preload tile 0 (async DMA into buffer 0)
    async_copy16(ldsA0, Ap0); async_copy16(ldsA1, Ap1);
    async_copy16(ldsB0, Bp0); async_copy16(ldsB1, Bp1);

    for (int kt = 0; kt < NT; ++kt) {
        wait_async0();          // my async copies into buf `cur` are in LDS
        __syncthreads();        // everyone's are
        const int cur = kt & 1;
        const unsigned nb = (unsigned)((kt + 1) & 1) * BUFB;

        // kick off DMA of the next tile into the other buffer
        if (kt + 1 < NT) {
            const int ko = (kt + 1) * 32;
            async_copy16(ldsA0 + nb, Ap0 + ko); async_copy16(ldsA1 + nb, Ap1 + ko);
            async_copy16(ldsB0 + nb, Bp0 + ko); async_copy16(ldsB1 + nb, Bp1 + ko);
        }

        // load all fragments, then issue all WMMAs
        Frag a[4], b[2];
        #pragma unroll
        for (int mf = 0; mf < 4; mf++) {
            const unsigned short* p = &As[cur][(waveM * 64 + mf * 16 + l15) * 40];
            a[mf].u[0] = *(const uint4*)&p[8 * hi];          // K 0..7   (+8*hi)
            a[mf].u[1] = *(const uint4*)&p[16 + 8 * hi];     // K 16..23 (+8*hi)
        }
        #pragma unroll
        for (int nf = 0; nf < 2; nf++) {
            const unsigned short* p = &Bs[cur][(waveN * 32 + nf * 16 + l15) * 40];
            b[nf].u[0] = *(const uint4*)&p[8 * hi];          // K 0..7   (+8*hi)
            b[nf].u[1] = *(const uint4*)&p[16 + 8 * hi];     // K 16..23 (+8*hi)
        }
        #pragma unroll
        for (int mf = 0; mf < 4; mf++)
            #pragma unroll
            for (int nf = 0; nf < 2; nf++)
                acc[mf][nf] = wmma_bf16(a[mf].v, b[nf].v, acc[mf][nf]);
    }

    // epilogue (lane = column n, vgpr j = row j + 8*hi within the 16x16 tile)
    #pragma unroll
    for (int mf = 0; mf < 4; mf++) {
        #pragma unroll
        for (int nf = 0; nf < 2; nf++) {
            const int n = nBase + waveN * 32 + nf * 16 + l15;
            const float bv = HAS_BIAS ? bias[n] : 0.f;
            #pragma unroll
            for (int j = 0; j < 8; j++) {
                const int m = mBase + waveM * 64 + mf * 16 + j + 8 * hi;
                float v = acc[mf][nf][j] + bv;
                if (RELU) v = fmaxf(v, 0.f);
                if (RES)  v += res[(size_t)m * N + n];
                if (OUT_BF16) ((unsigned short*)outp)[(size_t)m * N + n] = f2bf(v);
                else          ((float*)outp)[(size_t)m * N + n] = v;
            }
        }
    }
}

// ---------------------------------------------------------------------------
// Flash attention: block = 128 threads (4 waves), 64 query rows per block.
// grid = (S/64, B*H). Q/K/V/O are bf16 [B, S, D] with head h in cols h*64..
// K tile staged via async DMA (row-major, no transform); V staged manually
// (needs transpose); P re-laid-out through LDS.
// ---------------------------------------------------------------------------
__global__ __launch_bounds__(128) void attn_kernel(
    const unsigned short* __restrict__ Qp,
    const unsigned short* __restrict__ Kp,
    const unsigned short* __restrict__ Vp,
    unsigned short* __restrict__ Op)
{
    __shared__ __align__(16) unsigned short Ks [64 * 72];     // K rows, row-major
    __shared__ __align__(16) unsigned short VTs[64 * 72];     // V transposed: [d][k]
    __shared__ __align__(16) unsigned short Ps [4 * 16 * 72]; // per-wave P tiles

    const int tid = threadIdx.x, lane = tid & 31, wid = tid >> 5;
    const int l15 = lane & 15, hi = lane >> 4;
    const int qb  = blockIdx.x;
    const int bh  = blockIdx.y;
    const int bI  = bh >> 4, h = bh & 15;                     // H == 16
    const size_t base = (size_t)bI * SS * DD + (size_t)h * DK;
    const int qrow = qb * 64 + wid * 16 + l15;

    // thread-constant staging geometry
    const int kr0  = tid >> 3;            // K-tile row 0..15 (+16 per i)
    const int koff = (tid & 7) * 8;       // bf16 offset within 64-elem row
    const int vk0  = tid >> 5;            // V key 0..3 (+4 per i)
    const int vdp  = (tid & 31) * 2;      // even d
    const unsigned ldsK = (unsigned)(size_t)&Ks[kr0 * 72 + koff];

    // Q fragments (A layout), two k-steps covering d=0..63
    Frag qf[2];
    {
        const unsigned short* qp = Qp + base + (size_t)qrow * DD;
        #pragma unroll
        for (int t = 0; t < 2; t++) {
            qf[t].u[0] = *(const uint4*)&qp[t * 32 + 8 * hi];
            qf[t].u[1] = *(const uint4*)&qp[t * 32 + 16 + 8 * hi];
        }
    }

    f32x8 o[4];
    #pragma unroll
    for (int df = 0; df < 4; df++) o[df] = (f32x8){0.f,0.f,0.f,0.f,0.f,0.f,0.f,0.f};
    float mrun[8], lrun[8];
    #pragma unroll
    for (int j = 0; j < 8; j++) { mrun[j] = -1e30f; lrun[j] = 0.f; }

    for (int kb = 0; kb < SS; kb += 64) {
        // ---- stage K tile via async DMA (4 x 16B per thread)
        {
            const unsigned short* kp = Kp + base + (size_t)(kb + kr0) * DD + koff;
            #pragma unroll
            for (int i = 0; i < 4; i++)
                async_copy16(ldsK + (unsigned)(16 * i) * 144u,
                             kp + (size_t)(16 * i) * DD);
        }
        // ---- stage V transposed: issue all 16 loads, then all stores
        {
            unsigned int vreg[16];
            const unsigned short* vp = Vp + base + (size_t)(kb + vk0) * DD + vdp;
            #pragma unroll
            for (int i = 0; i < 16; i++)
                vreg[i] = *(const unsigned int*)(vp + (size_t)(4 * i) * DD);
            #pragma unroll
            for (int i = 0; i < 16; i++) {
                const int kk = vk0 + 4 * i;
                VTs[vdp * 72 + kk]       = (unsigned short)(vreg[i] & 0xffffu);
                VTs[(vdp + 1) * 72 + kk] = (unsigned short)(vreg[i] >> 16);
            }
        }
        wait_async0();
        __syncthreads();

        // ---- scores: load all 8 K fragments, then 8 WMMAs
        Frag bk[2][4];
        #pragma unroll
        for (int t = 0; t < 2; t++)
            #pragma unroll
            for (int nf = 0; nf < 4; nf++) {
                const unsigned short* p = &Ks[(nf * 16 + l15) * 72 + t * 32];
                bk[t][nf].u[0] = *(const uint4*)&p[8 * hi];
                bk[t][nf].u[1] = *(const uint4*)&p[16 + 8 * hi];
            }
        f32x8 s[4];
        #pragma unroll
        for (int nf = 0; nf < 4; nf++) s[nf] = (f32x8){0.f,0.f,0.f,0.f,0.f,0.f,0.f,0.f};
        #pragma unroll
        for (int t = 0; t < 2; t++)
            #pragma unroll
            for (int nf = 0; nf < 4; nf++)
                s[nf] = wmma_bf16(qf[t].v, bk[t][nf].v, s[nf]);

        #pragma unroll
        for (int nf = 0; nf < 4; nf++)
            #pragma unroll
            for (int j = 0; j < 8; j++)
                s[nf][j] *= 0.125f;                  // 1/sqrt(64)

        // ---- online softmax (row reductions across 16 lanes: lane = key col)
        #pragma unroll
        for (int j = 0; j < 8; j++) {
            float mx = -1e30f;
            #pragma unroll
            for (int nf = 0; nf < 4; nf++) mx = fmaxf(mx, s[nf][j]);
            #pragma unroll
            for (int off = 1; off < 16; off <<= 1) mx = fmaxf(mx, __shfl_xor(mx, off, 32));
            const float mnew = fmaxf(mrun[j], mx);
            const float corr = __expf(mrun[j] - mnew);
            float rsum = 0.f;
            #pragma unroll
            for (int nf = 0; nf < 4; nf++) {
                const float p = __expf(s[nf][j] - mnew);
                s[nf][j] = p;
                rsum += p;
            }
            #pragma unroll
            for (int off = 1; off < 16; off <<= 1) rsum += __shfl_xor(rsum, off, 32);
            lrun[j] = lrun[j] * corr + rsum;
            mrun[j] = mnew;
            #pragma unroll
            for (int df = 0; df < 4; df++) o[df][j] *= corr;
        }

        // ---- P (C-layout) -> LDS -> A-layout fragments
        #pragma unroll
        for (int nf = 0; nf < 4; nf++)
            #pragma unroll
            for (int j = 0; j < 8; j++)
                Ps[(wid * 16 + j + 8 * hi) * 72 + nf * 16 + l15] = f2bf(s[nf][j]);
        __syncthreads();

        // ---- O += P(16x64) @ V(64x64): load all fragments, then 8 WMMAs
        Frag pf[2], vf[2][4];
        #pragma unroll
        for (int t = 0; t < 2; t++) {
            const unsigned short* pp = &Ps[(wid * 16 + l15) * 72 + t * 32];
            pf[t].u[0] = *(const uint4*)&pp[8 * hi];
            pf[t].u[1] = *(const uint4*)&pp[16 + 8 * hi];
            #pragma unroll
            for (int df = 0; df < 4; df++) {
                const unsigned short* vp2 = &VTs[(df * 16 + l15) * 72 + t * 32];
                vf[t][df].u[0] = *(const uint4*)&vp2[8 * hi];
                vf[t][df].u[1] = *(const uint4*)&vp2[16 + 8 * hi];
            }
        }
        #pragma unroll
        for (int t = 0; t < 2; t++)
            #pragma unroll
            for (int df = 0; df < 4; df++)
                o[df] = wmma_bf16(pf[t].v, vf[t][df].v, o[df]);
        __syncthreads();
    }

    // finalize and store (bf16)
    #pragma unroll
    for (int j = 0; j < 8; j++) {
        const float invl = 1.0f / lrun[j];
        const int m = qb * 64 + wid * 16 + j + 8 * hi;
        #pragma unroll
        for (int df = 0; df < 4; df++)
            Op[base + (size_t)m * DD + df * 16 + l15] = f2bf(o[df][j] * invl);
    }
}

// ---------------------------------------------------------------------------
// Launch
// ---------------------------------------------------------------------------
extern "C" void kernel_launch(void* const* d_in, const int* in_sizes, int n_in,
                              void* d_out, int out_size, void* d_ws, size_t ws_size,
                              hipStream_t stream) {
    (void)in_sizes; (void)n_in; (void)out_size; (void)ws_size;

    const float* x     = (const float*)d_in[0];
    const float* ln1_g = (const float*)d_in[1];
    const float* ln1_b = (const float*)d_in[2];
    const float* wq    = (const float*)d_in[3];
    const float* wk    = (const float*)d_in[4];
    const float* wv    = (const float*)d_in[5];
    const float* wo    = (const float*)d_in[6];
    const float* ln2_g = (const float*)d_in[7];
    const float* ln2_b = (const float*)d_in[8];
    const float* w1    = (const float*)d_in[9];
    const float* b1    = (const float*)d_in[10];
    const float* w2    = (const float*)d_in[11];
    const float* b2    = (const float*)d_in[12];
    float* out = (float*)d_out;

    char* ws = (char*)d_ws;
    const size_t MB = 1ull << 20;
    // Workspace layout (80 MB total). hbuf aliases Qb/Kb/Vb/attn (all dead by
    // the time FFN1 runs); x2 aliases x1 (dead after QKV projections).
    unsigned short* wqT  = (unsigned short*)(ws + 0 * MB);   // 2 MB
    unsigned short* wkT  = (unsigned short*)(ws + 2 * MB);   // 2 MB
    unsigned short* wvT  = (unsigned short*)(ws + 4 * MB);   // 2 MB
    unsigned short* woT  = (unsigned short*)(ws + 6 * MB);   // 2 MB
    unsigned short* w1T  = (unsigned short*)(ws + 8 * MB);   // 8 MB  [FFN][D]
    unsigned short* w2T  = (unsigned short*)(ws + 16 * MB);  // 8 MB  [D][FFN]
    unsigned short* x1   = (unsigned short*)(ws + 24 * MB);  // 8 MB
    unsigned short* Qb   = (unsigned short*)(ws + 32 * MB);  // 8 MB
    unsigned short* Kb   = (unsigned short*)(ws + 40 * MB);  // 8 MB
    unsigned short* Vb   = (unsigned short*)(ws + 48 * MB);  // 8 MB
    unsigned short* attn = (unsigned short*)(ws + 56 * MB);  // 8 MB
    unsigned short* hbuf = (unsigned short*)(ws + 32 * MB);  // 32 MB (alias)
    float*          xres = (float*)(ws + 64 * MB);           // 16 MB
    unsigned short* x2   = x1;                               // alias

    // 1) weights -> bf16 transposed
    dim3 tb(32, 8);
    transpose_cvt_kernel<DD, DD><<<dim3(DD / 32, DD / 32), tb, 0, stream>>>(wq, wqT);
    transpose_cvt_kernel<DD, DD><<<dim3(DD / 32, DD / 32), tb, 0, stream>>>(wk, wkT);
    transpose_cvt_kernel<DD, DD><<<dim3(DD / 32, DD / 32), tb, 0, stream>>>(wv, wvT);
    transpose_cvt_kernel<DD, DD><<<dim3(DD / 32, DD / 32), tb, 0, stream>>>(wo, woT);
    transpose_cvt_kernel<DD, FFN><<<dim3(FFN / 32, DD / 32), tb, 0, stream>>>(w1, w1T);
    transpose_cvt_kernel<FFN, DD><<<dim3(DD / 32, FFN / 32), tb, 0, stream>>>(w2, w2T);

    // 2) LN1 -> bf16
    ln_bf16_kernel<<<MM, 256, 0, stream>>>(x, ln1_g, ln1_b, x1);

    // 3) Q, K, V projections
    dim3 g1(DD / 128, MM / 128);
    gemm_bf16_kernel<MM, DD, DD, false, false, false, true><<<g1, 256, 0, stream>>>(x1, wqT, nullptr, nullptr, Qb);
    gemm_bf16_kernel<MM, DD, DD, false, false, false, true><<<g1, 256, 0, stream>>>(x1, wkT, nullptr, nullptr, Kb);
    gemm_bf16_kernel<MM, DD, DD, false, false, false, true><<<g1, 256, 0, stream>>>(x1, wvT, nullptr, nullptr, Vb);

    // 4) attention
    attn_kernel<<<dim3(SS / 64, BB * HH), 128, 0, stream>>>(Qb, Kb, Vb, attn);

    // 5) output projection + residual -> xres (fp32)
    gemm_bf16_kernel<MM, DD, DD, false, false, true, false><<<g1, 256, 0, stream>>>(attn, woT, nullptr, x, xres);

    // 6) LN2 -> bf16
    ln_bf16_kernel<<<MM, 256, 0, stream>>>(xres, ln2_g, ln2_b, x2);

    // 7) FFN1: relu(x2 @ w1 + b1) -> bf16
    dim3 g2(FFN / 128, MM / 128);
    gemm_bf16_kernel<MM, FFN, DD, true, true, false, true><<<g2, 256, 0, stream>>>(x2, w1T, b1, nullptr, hbuf);

    // 8) FFN2: h @ w2 + b2 + xres -> d_out (fp32)
    gemm_bf16_kernel<MM, DD, FFN, true, false, true, false><<<g1, 256, 0, stream>>>(hbuf, w2T, b2, xres, out);
}
